// RecurrentModel_33157147525317
// MI455X (gfx1250) — matrix operand order repaired and checked by reading, not compiled, not run on hardware
//
#include <hip/hip_runtime.h>
#include <math.h>

// ---------------------------------------------------------------------------
// Seq2seq LSTM (2-layer encoder, 2-cell decoder) for MI455X / gfx1250.
// All GEMMs run on v_wmma_f32_16x16x32_f16 (f16 A/B, f32 accumulate) with
// 2x2 register blocking (32x32 output per wave) to double A/B fragment reuse.
// Recurrent steps are separate kernel launches on `stream` (graph-captured).
// ---------------------------------------------------------------------------

typedef __attribute__((ext_vector_type(16))) _Float16 v16h;
typedef __attribute__((ext_vector_type(8)))  float    v8f;

static constexpr int Bb   = 512;          // batch
static constexpr int Nn   = 128;          // feature dim
static constexpr int Tt   = 128;          // encoder steps
static constexpr int HORZ = 32;           // decoder horizon
static constexpr int HID  = 512;          // hidden
static constexpr int G4   = 4 * HID;      // gate width 2048

#define DEVI __device__ __forceinline__

// A-operand (16x32, f16) lane/slot mapping per CDNA5 ISA 7.12.2:
// lanes 0-15: M=0..15, VGPR0-3 hold K=0..7, VGPR4-7 hold K=16..23
// lanes16-31: M=0..15, VGPR0-3 hold K=8..15, VGPR4-7 hold K=24..31
DEVI size_t apack_index(int m, int k, int Ktot /* = K/32 */) {
    int tm = m >> 4, mm = m & 15;
    int kt = k >> 5, kk = k & 31;
    int lh   = (kk >> 3) & 1;                       // which lane half
    int lane = lh * 16 + mm;
    int v    = ((kk & 16) ? 4 : 0) + ((kk & 7) >> 1);
    int slot = v * 2 + (kk & 1);                    // half index 0..15
    return ((((size_t)tm * Ktot + kt) * 32) + lane) * 16 + slot;
}

// B-operand (32x16, f16): lanes 0-15 hold K=0..7,16..23, lanes16-31 K=8..15,24..31,
// column N = lane&15, same K order as the A operand.
DEVI size_t bpack_index(int n, int k, int Ktot /* = K/32 */) {
    int tn = n >> 4, nl = n & 15;
    int kt = k >> 5, kl = k & 31;
    int lane = ((kl >> 3) & 1) * 16 + nl;
    int slot = (kl & 7) + ((kl & 16) ? 8 : 0);
    return ((((size_t)tn * Ktot + kt) * 32) + lane) * 16 + slot;
}

DEVI float sigf(float x) { return 1.0f / (1.0f + __expf(-x)); }

// ---------------------------------------------------------------------------
// Utility kernels
// ---------------------------------------------------------------------------
__global__ void fill_f32(float* __restrict__ p, int n, float v) {
    int i = blockIdx.x * blockDim.x + threadIdx.x;
    if (i < n) p[i] = v;
}
__global__ void fill_f16(_Float16* __restrict__ p, int n) {
    int i = blockIdx.x * blockDim.x + threadIdx.x;
    if (i < n) p[i] = (_Float16)0.0f;
}
__global__ void add_bias(float* __restrict__ o, const float* __restrict__ a,
                         const float* __restrict__ b, int n) {
    int i = blockIdx.x * blockDim.x + threadIdx.x;
    if (i < n) o[i] = a[i] + b[i];
}
// W: [Ncols, K] row-major (torch weight). B-operand element (k,n) = W[n,k].
__global__ void pack_weight_b(const float* __restrict__ W, _Float16* __restrict__ out,
                              int Ncols, int K) {
    int i = blockIdx.x * blockDim.x + threadIdx.x;
    if (i >= Ncols * K) return;
    int n = i / K, k = i % K;
    out[bpack_index(n, k, K >> 5)] = (_Float16)W[i];
}
// window flat [B, T, N]  ->  xpack[t] in A-layout over (b, n), Ktot = N/32
__global__ void pack_window_a(const float* __restrict__ win, _Float16* __restrict__ xpack) {
    int i = blockIdx.x * blockDim.x + threadIdx.x;
    if (i >= Bb * Tt * Nn) return;
    int b = i / (Tt * Nn);
    int r = i - b * (Tt * Nn);
    int t = r / Nn, n = r - t * Nn;
    xpack[(size_t)t * (Bb * Nn) + apack_index(b, n, Nn >> 5)] = (_Float16)win[i];
}

// ---------------------------------------------------------------------------
// Dual-K WMMA GEMM, 2x2 register-blocked: each wave owns a 32x32 output block
// (tiles [tm0,tm1] x [tn0,tn1]).  D = A0*B0 + A1*B1 + bias.
//   A*: packed f16 A-layout  [Mtiles][kTiles][32 lanes][16 halfs]
//   B*: packed f16 B-layout  [Ntiles][kTiles][32 lanes][16 halfs]
// Epilogue (all optional): f32 gate buffer, repack-D-as-A-operand,
//   strided scatter into d_out.
// ---------------------------------------------------------------------------
struct EpiArgs {
    float*     gout;
    _Float16*  packout;
    int        packKtot;
    float*     scat;
    int        scatStride;
    int        Ncols;
};

DEVI void epilogue_tile(const v8f& acc, int tm, int tn, int lane, const EpiArgs& e) {
    int nn    = tn * 16 + (lane & 15);
    int mbase = tm * 16 + ((lane >> 4) << 3);
#pragma unroll
    for (int r = 0; r < 8; ++r) {
        int m = mbase + r;
        float vv = acc[r];
        if (e.gout)    e.gout[(size_t)m * e.Ncols + nn] = vv;
        if (e.packout) e.packout[apack_index(m, nn, e.packKtot)] = (_Float16)vv;
        if (e.scat)    e.scat[((size_t)m * e.Ncols + nn) * e.scatStride] = vv;
    }
}

__global__ void __launch_bounds__(256)
gemm_wmma_dual(const _Float16* __restrict__ A0, const _Float16* __restrict__ B0, int k0t,
               const _Float16* __restrict__ A1, const _Float16* __restrict__ B1, int k1t,
               const float* __restrict__ bias,
               float* __restrict__ gout,
               _Float16* __restrict__ packout, int packKtot,
               float* __restrict__ scat, int scatStride,
               int Mtiles, int Ntiles) {
    int MB = Mtiles >> 1, NB = Ntiles >> 1;        // 32x32 blocks
    int wave = blockIdx.x * (blockDim.x >> 5) + (threadIdx.x >> 5);
    int lane = threadIdx.x & 31;
    if (wave >= MB * NB) return;                   // grids are sized exact
    int bm = wave / NB, bn = wave - bm * NB;
    int tm0 = bm * 2, tm1 = tm0 + 1;
    int tn0 = bn * 2, tn1 = tn0 + 1;

    v8f acc00, acc01, acc10, acc11;
    float bv0 = bias ? bias[tn0 * 16 + (lane & 15)] : 0.0f;
    float bv1 = bias ? bias[tn1 * 16 + (lane & 15)] : 0.0f;
#pragma unroll
    for (int r = 0; r < 8; ++r) {
        acc00[r] = bv0; acc01[r] = bv1;
        acc10[r] = bv0; acc11[r] = bv1;
    }

    {
        const _Float16* Ap0 = A0 + ((size_t)tm0 * k0t) * 512 + lane * 16;
        const _Float16* Ap1 = A0 + ((size_t)tm1 * k0t) * 512 + lane * 16;
        const _Float16* Bp0 = B0 + ((size_t)tn0 * k0t) * 512 + lane * 16;
        const _Float16* Bp1 = B0 + ((size_t)tn1 * k0t) * 512 + lane * 16;
        for (int kt = 0; kt < k0t; ++kt) {
            v16h a0 = *(const v16h*)(Ap0 + (size_t)kt * 512);
            v16h a1 = *(const v16h*)(Ap1 + (size_t)kt * 512);
            v16h b0 = *(const v16h*)(Bp0 + (size_t)kt * 512);
            v16h b1 = *(const v16h*)(Bp1 + (size_t)kt * 512);
            acc00 = __builtin_amdgcn_wmma_f32_16x16x32_f16(false, a0, false, b0, (short)0, acc00, false, false);
            acc01 = __builtin_amdgcn_wmma_f32_16x16x32_f16(false, a0, false, b1, (short)0, acc01, false, false);
            acc10 = __builtin_amdgcn_wmma_f32_16x16x32_f16(false, a1, false, b0, (short)0, acc10, false, false);
            acc11 = __builtin_amdgcn_wmma_f32_16x16x32_f16(false, a1, false, b1, (short)0, acc11, false, false);
        }
    }
    if (A1) {
        const _Float16* Ap0 = A1 + ((size_t)tm0 * k1t) * 512 + lane * 16;
        const _Float16* Ap1 = A1 + ((size_t)tm1 * k1t) * 512 + lane * 16;
        const _Float16* Bp0 = B1 + ((size_t)tn0 * k1t) * 512 + lane * 16;
        const _Float16* Bp1 = B1 + ((size_t)tn1 * k1t) * 512 + lane * 16;
        for (int kt = 0; kt < k1t; ++kt) {
            v16h a0 = *(const v16h*)(Ap0 + (size_t)kt * 512);
            v16h a1 = *(const v16h*)(Ap1 + (size_t)kt * 512);
            v16h b0 = *(const v16h*)(Bp0 + (size_t)kt * 512);
            v16h b1 = *(const v16h*)(Bp1 + (size_t)kt * 512);
            acc00 = __builtin_amdgcn_wmma_f32_16x16x32_f16(false, a0, false, b0, (short)0, acc00, false, false);
            acc01 = __builtin_amdgcn_wmma_f32_16x16x32_f16(false, a0, false, b1, (short)0, acc01, false, false);
            acc10 = __builtin_amdgcn_wmma_f32_16x16x32_f16(false, a1, false, b0, (short)0, acc10, false, false);
            acc11 = __builtin_amdgcn_wmma_f32_16x16x32_f16(false, a1, false, b1, (short)0, acc11, false, false);
        }
    }

    EpiArgs e{gout, packout, packKtot, scat, scatStride, Ntiles * 16};
    epilogue_tile(acc00, tm0, tn0, lane, e);
    epilogue_tile(acc01, tm0, tn1, lane, e);
    epilogue_tile(acc10, tm1, tn0, lane, e);
    epilogue_tile(acc11, tm1, tn1, lane, e);
}

// ---------------------------------------------------------------------------
// Pointwise LSTM cell update (torch gate order i,f,g,o).
// Writes new c (f32) and new h directly in packed A-operand f16 layout.
// Optional: post-cell h/c dropout masks; optional masked-h side store
// (encoder inter-layer dropout) also in A-operand layout.
// ---------------------------------------------------------------------------
__global__ void __launch_bounds__(256)
lstm_update(const float* __restrict__ g, float* __restrict__ c,
            _Float16* __restrict__ hpack,
            const float* __restrict__ hmask, const float* __restrict__ cmask,
            const float* __restrict__ smask, _Float16* __restrict__ spack, int n) {
    int i = blockIdx.x * blockDim.x + threadIdx.x;
    if (i >= n) return;
    int b = i >> 9;            // / HID
    int j = i & (HID - 1);
    const float* gr = g + (size_t)b * G4;
    float gi = gr[j], gf = gr[HID + j], gg = gr[2 * HID + j], go = gr[3 * HID + j];
    float cn = sigf(gf) * c[i] + sigf(gi) * tanhf(gg);
    float hn = sigf(go) * tanhf(cn);
    if (smask) spack[apack_index(b, j, HID >> 5)] = (_Float16)(hn * smask[i]);
    if (hmask) hn *= hmask[i];
    if (cmask) cn *= cmask[i];
    c[i] = cn;
    hpack[apack_index(b, j, HID >> 5)] = (_Float16)hn;
}

// ---------------------------------------------------------------------------
extern "C" void kernel_launch(void* const* d_in, const int* in_sizes, int n_in,
                              void* d_out, int out_size, void* d_ws, size_t ws_size,
                              hipStream_t stream) {
    (void)in_sizes; (void)n_in; (void)out_size; (void)ws_size;
    const float* window   = (const float*)d_in[0];
    const float* Wih0 = (const float*)d_in[1];  const float* Whh0 = (const float*)d_in[2];
    const float* bih0 = (const float*)d_in[3];  const float* bhh0 = (const float*)d_in[4];
    const float* Wih1 = (const float*)d_in[5];  const float* Whh1 = (const float*)d_in[6];
    const float* bih1 = (const float*)d_in[7];  const float* bhh1 = (const float*)d_in[8];
    const float* Wc1i = (const float*)d_in[9];  const float* Wc1h = (const float*)d_in[10];
    const float* bc1i = (const float*)d_in[11]; const float* bc1h = (const float*)d_in[12];
    const float* Wc2i = (const float*)d_in[13]; const float* Wc2h = (const float*)d_in[14];
    const float* bc2i = (const float*)d_in[15]; const float* bc2h = (const float*)d_in[16];
    const float* Wp   = (const float*)d_in[17]; const float* bp   = (const float*)d_in[18];
    const float* enc_mask   = (const float*)d_in[19];
    const float* dec_h_mask = (const float*)d_in[20];
    const float* dec_c_mask = (const float*)d_in[21];
    float* out = (float*)d_out;

    // ---- workspace carve-out (256B aligned) ----
    char*  ws  = (char*)d_ws;
    size_t off = 0;
    auto alloc = [&](size_t bytes) -> char* {
        off = (off + 255) & ~(size_t)255;
        char* p = ws + off;
        off += bytes;
        return p;
    };
    auto h16 = [&](size_t nel) { return (_Float16*)alloc(nel * sizeof(_Float16)); };
    auto f32 = [&](size_t nel) { return (float*)alloc(nel * sizeof(float)); };

    _Float16* WihB0 = h16((size_t)G4 * Nn);
    _Float16* WhhB0 = h16((size_t)G4 * HID);
    _Float16* WihB1 = h16((size_t)G4 * HID);
    _Float16* WhhB1 = h16((size_t)G4 * HID);
    _Float16* Wc1iB = h16((size_t)G4 * Nn);
    _Float16* Wc1hB = h16((size_t)G4 * HID);
    _Float16* Wc2iB = h16((size_t)G4 * HID);
    _Float16* Wc2hB = h16((size_t)G4 * HID);
    _Float16* WpB   = h16((size_t)Nn * HID);
    _Float16* xpack   = h16((size_t)Tt * Bb * Nn);    // 16 MB
    _Float16* hs0pack = h16((size_t)Tt * Bb * HID);   // 64 MB
    _Float16* hApack  = h16((size_t)Bb * HID);
    _Float16* hBpack  = h16((size_t)Bb * HID);
    _Float16* predpk  = h16((size_t)Bb * Nn);
    float* gbuf  = f32((size_t)Bb * G4);              // 4 MB
    float* cA    = f32((size_t)Bb * HID);
    float* cB    = f32((size_t)Bb * HID);
    float* biasE0 = f32(G4); float* biasE1 = f32(G4);
    float* biasC1 = f32(G4); float* biasC2 = f32(G4);

    const int BH  = Bb * HID;                 // 262144
    const int UPD = BH / 256;                 // 1024 blocks
    auto grid1 = [](int n) { return (n + 255) / 256; };

    // ---- one-time prep (per call; deterministic) ----
    pack_weight_b<<<grid1(G4 * Nn), 256, 0, stream>>>(Wih0, WihB0, G4, Nn);
    pack_weight_b<<<grid1(G4 * HID), 256, 0, stream>>>(Whh0, WhhB0, G4, HID);
    pack_weight_b<<<grid1(G4 * HID), 256, 0, stream>>>(Wih1, WihB1, G4, HID);
    pack_weight_b<<<grid1(G4 * HID), 256, 0, stream>>>(Whh1, WhhB1, G4, HID);
    pack_weight_b<<<grid1(G4 * Nn), 256, 0, stream>>>(Wc1i, Wc1iB, G4, Nn);
    pack_weight_b<<<grid1(G4 * HID), 256, 0, stream>>>(Wc1h, Wc1hB, G4, HID);
    pack_weight_b<<<grid1(G4 * HID), 256, 0, stream>>>(Wc2i, Wc2iB, G4, HID);
    pack_weight_b<<<grid1(G4 * HID), 256, 0, stream>>>(Wc2h, Wc2hB, G4, HID);
    pack_weight_b<<<grid1(Nn * HID), 256, 0, stream>>>(Wp, WpB, Nn, HID);
    add_bias<<<grid1(G4), 256, 0, stream>>>(biasE0, bih0, bhh0, G4);
    add_bias<<<grid1(G4), 256, 0, stream>>>(biasE1, bih1, bhh1, G4);
    add_bias<<<grid1(G4), 256, 0, stream>>>(biasC1, bc1i, bc1h, G4);
    add_bias<<<grid1(G4), 256, 0, stream>>>(biasC2, bc2i, bc2h, G4);
    pack_window_a<<<grid1(Bb * Tt * Nn), 256, 0, stream>>>(window, xpack);
    fill_f32<<<UPD, 256, 0, stream>>>(cA, BH, 0.0f);
    fill_f32<<<UPD, 256, 0, stream>>>(cB, BH, 0.0f);
    fill_f16<<<UPD, 256, 0, stream>>>(hApack, BH);
    fill_f16<<<UPD, 256, 0, stream>>>(hBpack, BH);

    const int MT = Bb / 16;                   // 32 M-tiles
    const int NT = G4 / 16;                   // 128 N-tiles (gates)
    const int GEMM_BLK = (MT / 2) * (NT / 2) / 8;   // 128 blocks, 8 waves each
    const int PNT = Nn / 16;                  // 8 N-tiles (projection)
    const int PRED_BLK = (MT / 2) * (PNT / 2) / 8;  // 8 blocks

    // ---- encoder layer 0 ----
    for (int t = 0; t < Tt; ++t) {
        gemm_wmma_dual<<<GEMM_BLK, 256, 0, stream>>>(
            xpack + (size_t)t * Bb * Nn, WihB0, Nn / 32,
            hApack, WhhB0, HID / 32, biasE0,
            gbuf, nullptr, 0, nullptr, 0, MT, NT);
        lstm_update<<<UPD, 256, 0, stream>>>(
            gbuf, cA, hApack, nullptr, nullptr,
            enc_mask + (size_t)t * BH, hs0pack + (size_t)t * BH, BH);
    }
    // ---- encoder layer 1 (consumes masked layer-0 outputs) ----
    for (int t = 0; t < Tt; ++t) {
        gemm_wmma_dual<<<GEMM_BLK, 256, 0, stream>>>(
            hs0pack + (size_t)t * BH, WihB1, HID / 32,
            hBpack, WhhB1, HID / 32, biasE1,
            gbuf, nullptr, 0, nullptr, 0, MT, NT);
        lstm_update<<<UPD, 256, 0, stream>>>(
            gbuf, cB, hBpack, nullptr, nullptr, nullptr, nullptr, BH);
    }
    // ---- decoder ----
    for (int t = 0; t < HORZ - 1; ++t) {
        // pred = h2 @ Wp^T + bp ; scatter to out[:, t]; repack as A operand
        gemm_wmma_dual<<<PRED_BLK, 256, 0, stream>>>(
            hBpack, WpB, HID / 32, nullptr, nullptr, 0, bp,
            nullptr, predpk, Nn / 32, out + t, HORZ, MT, PNT);
        // cell1: g = pred@Wc1i^T + h1@Wc1h^T + b
        gemm_wmma_dual<<<GEMM_BLK, 256, 0, stream>>>(
            predpk, Wc1iB, Nn / 32, hApack, Wc1hB, HID / 32, biasC1,
            gbuf, nullptr, 0, nullptr, 0, MT, NT);
        lstm_update<<<UPD, 256, 0, stream>>>(
            gbuf, cA, hApack,
            dec_h_mask + (size_t)t * BH, dec_c_mask + (size_t)t * BH,
            nullptr, nullptr, BH);
        // cell2: g = h1@Wc2i^T + h2@Wc2h^T + b
        gemm_wmma_dual<<<GEMM_BLK, 256, 0, stream>>>(
            hApack, Wc2iB, HID / 32, hBpack, Wc2hB, HID / 32, biasC2,
            gbuf, nullptr, 0, nullptr, 0, MT, NT);
        lstm_update<<<UPD, 256, 0, stream>>>(
            gbuf, cB, hBpack, nullptr, nullptr, nullptr, nullptr, BH);
    }
    // final projection -> out[:, HOR-1]
    gemm_wmma_dual<<<PRED_BLK, 256, 0, stream>>>(
        hBpack, WpB, HID / 32, nullptr, nullptr, 0, bp,
        nullptr, nullptr, 0, out + (HORZ - 1), HORZ, MT, PNT);
}
